// KNN_32220844654874
// MI455X (gfx1250) — matrix operand order repaired and checked by reading, not compiled, not run on hardware
//
#include <hip/hip_runtime.h>
#include <hip/hip_bf16.h>
#include <stdint.h>

// ---------------------------------------------------------------------------
// KNN (1-nearest-neighbor) for B=1024 queries vs N=100000 train rows, D=768.
// Strategy: bf16 hi/lo split-precision GEMM on v_wmma_f32_16x16x32_bf16
// (3 WMMAs emulate fp32 dot products), fused distance + per-chunk argmin,
// then a cross-chunk reduction + gather of Y_train[nearest].
// ---------------------------------------------------------------------------

typedef __attribute__((ext_vector_type(16))) __bf16          v16bf;
typedef __attribute__((ext_vector_type(16))) unsigned short  v16us;
typedef __attribute__((ext_vector_type(8)))  float           v8f;

#define D_DIM   768
#define N_TRAIN 100000
#define B_Q     1024
#define BM      128      // query rows per block (8 waves x 16)
#define BN      128      // train rows per block (8 tiles of 16)
#define KT      32       // K per WMMA step (bf16)
#define YDIM    24
#define BIGF    3.0e38f

__device__ __forceinline__ unsigned short bf_bits(__bf16 h) {
  return __builtin_bit_cast(unsigned short, h);
}

// A-matrix (16x32 bf16) fragment mapping, per ISA 7.12.2:
// lane = (m&15) | (((k>>3)&1)<<4) ; vgpr v = ((k&16)>>2) + ((k>>1)&3) ; elem = 2v + (k&1)
__device__ __forceinline__ void amap(int k, int& laneOff, int& e) {
  laneOff = ((k >> 3) & 1) << 4;
  int v = ((k & 16) >> 2) + ((k >> 1) & 3);
  e = 2 * v + (k & 1);
}

// B-matrix (32x16 bf16) fragment mapping (per 64x16 pattern in ISA):
// lane = (n&15) | (((k>>3)&1)<<4) ; vgpr v = ((k&16)>>2) + ((k>>1)&3) ; elem = 2v + (k&1)
__device__ __forceinline__ void bmap(int k, int& laneOff, int& e) {
  laneOff = ((k >> 3) & 1) << 4;
  int v = ((k & 16) >> 2) + ((k >> 1) & 3);
  e = 2 * v + (k & 1);
}

__device__ __forceinline__ v16bf load_frag(const unsigned short* p) {
  v16us u = *(const v16us*)p;             // 32B contiguous per lane in LDS
  return __builtin_bit_cast(v16bf, u);
}

// ---------------------------------------------------------------------------
// Row sum-of-squares: one wave per row.
// ---------------------------------------------------------------------------
__global__ __launch_bounds__(256) void rowsumsq(const float* __restrict__ X,
                                                float* __restrict__ out,
                                                int rows, int cols) {
  int row  = blockIdx.x * 8 + (threadIdx.x >> 5);
  int lane = threadIdx.x & 31;
  if (row >= rows) return;
  const float* p = X + (size_t)row * cols;
  float s = 0.f;
  for (int c = lane; c < cols; c += 32) { float v = p[c]; s += v * v; }
  #pragma unroll
  for (int off = 16; off >= 1; off >>= 1) s += __shfl_xor(s, off, 32);
  if (lane == 0) out[row] = s;
}

// ---------------------------------------------------------------------------
// Main fused GEMM + distance + per-chunk argmin.
// Grid: (nChunks, B/BM). Block: 256 threads (8 waves).
// Each wave computes a 16(query) x 128(train) cross tile in 8 v8f accumulators.
// ---------------------------------------------------------------------------
__global__ __launch_bounds__(256) void knn_gemm_partial(
    const float* __restrict__ Xq, const float* __restrict__ Xt,
    const float* __restrict__ q2, const float* __restrict__ t2,
    float* __restrict__ pmin, int* __restrict__ pidx, int nChunks)
{
  __shared__ __align__(32) unsigned short ldsQ[2][8][32][16];  // [hi/lo][rowblk][lane][elem]
  __shared__ __align__(32) unsigned short ldsT[2][8][32][16];

  const int chunk = blockIdx.x;
  const int btile = blockIdx.y;
  const int n0 = chunk * BN;
  const int m0 = btile * BM;
  const int tid  = threadIdx.x;
  const int lane = tid & 31;
  const int wave = tid >> 5;

  v8f acc[8];
  #pragma unroll
  for (int i = 0; i < 8; ++i) acc[i] = (v8f){};

  // Cooperative loader assignment: thread -> (local row, k half)
  const int lrow = tid >> 1;           // 0..127
  const int k0   = (tid & 1) * 16;     // 0 or 16
  const int rb   = lrow >> 4;
  const int mlan = lrow & 15;
  const int tRow = n0 + lrow;
  const bool tValid = (tRow < N_TRAIN);

  const float* qsrcRow = Xq + (size_t)(m0 + lrow) * D_DIM + k0;
  const float* tsrcRow = Xt + (size_t)(tValid ? tRow : 0) * D_DIM + k0;

  for (int ks = 0; ks < D_DIM; ks += KT) {
    // ---- stage Q tile (128 x 32) as bf16 hi/lo into A-fragment layout ----
    {
      const float* src = qsrcRow + ks;
      #pragma unroll
      for (int i = 0; i < 16; i += 2) {
        float f0 = src[i], f1 = src[i + 1];
        __bf16 h0 = (__bf16)f0, h1 = (__bf16)f1;
        __bf16 l0 = (__bf16)(f0 - (float)h0), l1 = (__bf16)(f1 - (float)h1);
        int k = k0 + i, lo, e;
        amap(k, lo, e);
        int lt = mlan | lo;
        unsigned int ph = (unsigned int)bf_bits(h0) | ((unsigned int)bf_bits(h1) << 16);
        unsigned int pl = (unsigned int)bf_bits(l0) | ((unsigned int)bf_bits(l1) << 16);
        *(unsigned int*)&ldsQ[0][rb][lt][e] = ph;
        *(unsigned int*)&ldsQ[1][rb][lt][e] = pl;
      }
    }
    // ---- stage T tile (128 x 32) as bf16 hi/lo into B-fragment layout ----
    {
      const float* src = tsrcRow + ks;
      #pragma unroll
      for (int i = 0; i < 16; i += 2) {
        float f0 = tValid ? src[i] : 0.f;
        float f1 = tValid ? src[i + 1] : 0.f;
        __bf16 h0 = (__bf16)f0, h1 = (__bf16)f1;
        __bf16 l0 = (__bf16)(f0 - (float)h0), l1 = (__bf16)(f1 - (float)h1);
        int k = k0 + i, lo, e;
        bmap(k, lo, e);
        int lt = mlan | lo;
        unsigned int ph = (unsigned int)bf_bits(h0) | ((unsigned int)bf_bits(h1) << 16);
        unsigned int pl = (unsigned int)bf_bits(l0) | ((unsigned int)bf_bits(l1) << 16);
        *(unsigned int*)&ldsT[0][rb][lt][e] = ph;
        *(unsigned int*)&ldsT[1][rb][lt][e] = pl;
      }
    }
    __syncthreads();

    // ---- WMMA: cross += ah*bh + ah*bl + al*bh (split-precision fp32) ----
    v16bf ah = load_frag(&ldsQ[0][wave][lane][0]);
    v16bf al = load_frag(&ldsQ[1][wave][lane][0]);
    #pragma unroll
    for (int nt = 0; nt < 8; ++nt) {
      v16bf bh = load_frag(&ldsT[0][nt][lane][0]);
      v16bf bl = load_frag(&ldsT[1][nt][lane][0]);
      acc[nt] = __builtin_amdgcn_wmma_f32_16x16x32_bf16(false, ah, false, bh,
                                                        (short)0, acc[nt], false, false);
      acc[nt] = __builtin_amdgcn_wmma_f32_16x16x32_bf16(false, ah, false, bl,
                                                        (short)0, acc[nt], false, false);
      acc[nt] = __builtin_amdgcn_wmma_f32_16x16x32_bf16(false, al, false, bh,
                                                        (short)0, acc[nt], false, false);
    }
    __syncthreads();
  }

  // ---- epilogue: d2 = q2 + t2 - 2*cross ; per-query argmin over this chunk ----
  const int half = lane >> 4;
  const int nl   = lane & 15;

  float tv[8];
  #pragma unroll
  for (int nt = 0; nt < 8; ++nt) {
    int n = n0 + nt * 16 + nl;
    tv[nt] = (n < N_TRAIN) ? t2[n] : BIGF;
  }

  #pragma unroll
  for (int r = 0; r < 8; ++r) {
    int m = m0 + wave * 16 + r + 8 * half;   // C/D layout: M = r + 8*(lane>>4)
    float q2v = q2[m];
    float best = BIGF;
    int   bidx = 0x7fffffff;
    #pragma unroll
    for (int nt = 0; nt < 8; ++nt) {
      int n = n0 + nt * 16 + nl;
      float d2 = q2v + tv[nt] - 2.0f * acc[nt][r];
      if (n >= N_TRAIN) d2 = BIGF;
      if (d2 < best || (d2 == best && n < bidx)) { best = d2; bidx = n; }
    }
    // reduce across the 16 lanes of this half-wave (xor keeps bit4 fixed)
    #pragma unroll
    for (int off = 8; off >= 1; off >>= 1) {
      float ob = __shfl_xor(best, off, 32);
      int   oi = __shfl_xor(bidx, off, 32);
      if (ob < best || (ob == best && oi < bidx)) { best = ob; bidx = oi; }
    }
    if (nl == 0) {
      pmin[(size_t)m * nChunks + chunk] = best;
      pidx[(size_t)m * nChunks + chunk] = bidx;
    }
  }
}

// ---------------------------------------------------------------------------
// Final reduction over chunks + gather Y_train[nearest] -> out[B,24]
// ---------------------------------------------------------------------------
__global__ __launch_bounds__(256) void knn_finalize(
    const float* __restrict__ pmin, const int* __restrict__ pidx,
    const float* __restrict__ Y, float* __restrict__ out, int nChunks)
{
  int q = blockIdx.x * blockDim.x + threadIdx.x;
  if (q >= B_Q) return;
  const float* pm = pmin + (size_t)q * nChunks;
  const int*   pi = pidx + (size_t)q * nChunks;
  float best = BIGF;
  int   bidx = 0;
  for (int c = 0; c < nChunks; ++c) {
    float v = pm[c];
    int   i = pi[c];
    if (v < best || (v == best && i < bidx)) { best = v; bidx = i; }
  }
  const float* y = Y + (size_t)bidx * YDIM;
  float*       o = out + (size_t)q * YDIM;
  #pragma unroll
  for (int j = 0; j < YDIM; ++j) o[j] = y[j];
}

// ---------------------------------------------------------------------------
extern "C" void kernel_launch(void* const* d_in, const int* in_sizes, int n_in,
                              void* d_out, int out_size, void* d_ws, size_t ws_size,
                              hipStream_t stream) {
  const float* x  = (const float*)d_in[0];   // [1024, 48, 16] -> [1024, 768]
  const float* Xt = (const float*)d_in[1];   // [100000, 768]
  const float* Yt = (const float*)d_in[2];   // [100000, 24, 1]
  float* out = (float*)d_out;                // [1024, 24]

  const int nChunks = (N_TRAIN + BN - 1) / BN;   // 782

  // workspace layout
  char* ws = (char*)d_ws;
  float* q2   = (float*)(ws);                          //   4 KB reserved
  float* t2   = (float*)(ws + 4096);                   // 400000 B
  float* pmin = (float*)(ws + 404096);                 // 1024*782*4 = 3203072 B
  int*   pidx = (int*)  (ws + 404096 + 3203072);       // 3203072 B

  rowsumsq<<<B_Q / 8, 256, 0, stream>>>(x, q2, B_Q, D_DIM);
  rowsumsq<<<(N_TRAIN + 7) / 8, 256, 0, stream>>>(Xt, t2, N_TRAIN, D_DIM);

  dim3 grid(nChunks, B_Q / BM);
  knn_gemm_partial<<<grid, 256, 0, stream>>>(x, Xt, q2, t2, pmin, pidx, nChunks);

  knn_finalize<<<(B_Q + 255) / 256, 256, 0, stream>>>(pmin, pidx, Yt, out, nChunks);
}
